// SAModule_5454608466697
// MI455X (gfx1250) — matrix operand order repaired and checked by reading, not compiled, not run on hardware
//
#include <hip/hip_runtime.h>

typedef __attribute__((ext_vector_type(16))) _Float16 v16h;
typedef __attribute__((ext_vector_type(8)))  _Float16 v8h;
typedef __attribute__((ext_vector_type(2)))  _Float16 h2;
typedef __attribute__((ext_vector_type(8)))  float    v8f;
typedef __attribute__((ext_vector_type(4)))  unsigned int u32x4;
typedef __attribute__((ext_vector_type(4)))  int      i32x4;
typedef __attribute__((ext_vector_type(8)))  int      i32x8;

constexpr int NB = 16;     // batch
constexpr int NN = 4096;   // points
constexpr int NC = 64;     // channels
constexpr int NM = 1024;   // sampled points (NN/4)
constexpr int NK = 32;     // kNN
constexpr int KP = 128;    // padded inner dim (67 -> 128)
constexpr int H3 = 256;    // output channels
constexpr int WAVES = 8;   // waves per workgroup (mlp kernel)
constexpr int CPW   = 8;   // centroids per wave

// weight image: [128][128] + [128][128] + [256][128] f16, then 512 f32 biases
constexpr int WP_HALVES = 128 * KP + 128 * KP + H3 * KP;         // 65536
constexpr int WP_BYTES  = WP_HALVES * 2 + 512 * 4;               // 133120
constexpr int WP_QW     = WP_BYTES / 8;                          // 16640 (8B elems)

// squared term with rounding barrier: prevents fma contraction so arithmetic
// matches XLA's (dx*dx + dy*dy) + dz*dz elementwise-square-then-sum order.
__device__ __forceinline__ float sq_nofma(float d) {
  float r = d * d;
  asm volatile("" : "+v"(r));
  return r;
}

// ---------------------------------------------------------------------------
// Kernel 1: farthest point sampling. One block per cloud, 256 threads,
// 16 points/thread held entirely in registers. Exact f32, first-occurrence
// argmax tie-break (smaller index wins on equal distance).
// ---------------------------------------------------------------------------
__global__ __launch_bounds__(256) void fps_kernel(const float* __restrict__ pos,
                                                  float* __restrict__ poss) {
  __shared__ float s_c[3];
  __shared__ float s_rv[8];
  __shared__ int   s_ri[8];
  __shared__ int   s_next;

  const int b    = blockIdx.x;
  const int tid  = threadIdx.x;
  const int lane = tid & 31;
  const int w    = tid >> 5;

  float px[16], py[16], pz[16], md[16];
#pragma unroll
  for (int i = 0; i < 16; ++i) {
    const float* p = pos + ((size_t)b * NN + (i * 256 + tid)) * 3;
    px[i] = p[0]; py[i] = p[1]; pz[i] = p[2];
    md[i] = 1e30f;
  }

  int cur = 0;
  for (int m = 0; m < NM; ++m) {
#pragma unroll
    for (int i = 0; i < 16; ++i)
      if (cur == i * 256 + tid) { s_c[0] = px[i]; s_c[1] = py[i]; s_c[2] = pz[i]; }
    __syncthreads();

    if (tid == 0) {
      float* pr = poss + ((size_t)b * NM + m) * 3;
      pr[0] = s_c[0]; pr[1] = s_c[1]; pr[2] = s_c[2];
    }
    const float cx = s_c[0], cy = s_c[1], cz = s_c[2];

    float bv = -1.0f;
    int   bn = 0x7fffffff;
#pragma unroll
    for (int i = 0; i < 16; ++i) {
      const float d = (sq_nofma(px[i] - cx) + sq_nofma(py[i] - cy)) + sq_nofma(pz[i] - cz);
      md[i] = fminf(md[i], d);
      const int n = i * 256 + tid;
      if (md[i] > bv || (md[i] == bv && n < bn)) { bv = md[i]; bn = n; }
    }
#pragma unroll
    for (int off = 16; off >= 1; off >>= 1) {
      const float ov = __shfl_xor(bv, off);
      const int   on = __shfl_xor(bn, off);
      if (ov > bv || (ov == bv && on < bn)) { bv = ov; bn = on; }
    }
    if (lane == 0) { s_rv[w] = bv; s_ri[w] = bn; }
    __syncthreads();
    if (tid == 0) {
      float xv = s_rv[0]; int xn = s_ri[0];
#pragma unroll
      for (int q = 1; q < 8; ++q)
        if (s_rv[q] > xv || (s_rv[q] == xv && s_ri[q] < xn)) { xv = s_rv[q]; xn = s_ri[q]; }
      s_next = xn;
    }
    __syncthreads();
    cur = s_next;
  }
}

// ---------------------------------------------------------------------------
// Kernel 2: kNN (K=32) per centroid; register-resident candidate list.
// ---------------------------------------------------------------------------
__global__ __launch_bounds__(256) void knn_kernel(const float* __restrict__ pos,
                                                  const float* __restrict__ poss,
                                                  int* __restrict__ knn) {
  __shared__ float sp[256 * 3];
  const int tid = threadIdx.x;
  const int c   = blockIdx.x * 256 + tid;   // whole block shares one batch b
  const int b   = c >> 10;

  const float cx = poss[(size_t)c * 3 + 0];
  const float cy = poss[(size_t)c * 3 + 1];
  const float cz = poss[(size_t)c * 3 + 2];

  float dist[NK]; int nbr[NK];
#pragma unroll
  for (int k = 0; k < NK; ++k) { dist[k] = 3.0e38f; nbr[k] = 0; }
  float worst = 3.0e38f;

  const float* pb = pos + (size_t)b * NN * 3;
  for (int t0 = 0; t0 < NN; t0 += 256) {
    __syncthreads();
    for (int q = tid; q < 768; q += 256) sp[q] = pb[(size_t)t0 * 3 + q];
    __syncthreads();
    for (int jj = 0; jj < 256; ++jj) {
      const float d = (sq_nofma(sp[jj * 3 + 0] - cx) + sq_nofma(sp[jj * 3 + 1] - cy))
                      + sq_nofma(sp[jj * 3 + 2] - cz);
      if (d < worst) {
        int wp = 0; float wv = dist[0];
#pragma unroll
        for (int k = 1; k < NK; ++k) if (dist[k] > wv) { wv = dist[k]; wp = k; }
#pragma unroll
        for (int k = 0; k < NK; ++k) if (k == wp) { dist[k] = d; nbr[k] = t0 + jj; }
        wv = dist[0];
#pragma unroll
        for (int k = 1; k < NK; ++k) wv = fmaxf(wv, dist[k]);
        worst = wv;
      }
    }
  }
#pragma unroll
  for (int k = 0; k < NK; ++k) knn[(size_t)c * NK + k] = nbr[k];
}

// ---------------------------------------------------------------------------
// Prep kernels: build the contiguous f16 weight image (transposed, padded,
// + f32 biases) and the f16 copy of x, both in workspace. Done once per call;
// every mlp workgroup then DMAs the weight image via one TDM descriptor.
// ---------------------------------------------------------------------------
__global__ __launch_bounds__(256) void prep_weights(
    const float* __restrict__ W1, const float* __restrict__ b1,
    const float* __restrict__ W2, const float* __restrict__ b2,
    const float* __restrict__ W3, const float* __restrict__ b3,
    _Float16* __restrict__ wp) {
  const int i = blockIdx.x * 256 + threadIdx.x;   // 0..65535
  _Float16 v;
  if (i < 16384) {                 // W1t[n][k] = W1[k][n], k>=67 zero-padded
    const int n = i >> 7, k = i & 127;
    v = (_Float16)((k < NC + 3) ? W1[k * 128 + n] : 0.0f);
  } else if (i < 32768) {          // W2t
    const int j = i - 16384; const int n = j >> 7, k = j & 127;
    v = (_Float16)W2[k * 128 + n];
  } else {                         // W3t
    const int j = i - 32768; const int n = j >> 7, k = j & 127;
    v = (_Float16)W3[k * H3 + n];
  }
  wp[i] = v;
  float* bp = (float*)(wp + WP_HALVES);
  if (i < 512) bp[i] = (i < 128) ? b1[i] : (i < 256) ? b2[i - 128] : b3[i - 256];
}

__global__ __launch_bounds__(256) void prep_x(const float* __restrict__ x,
                                              _Float16* __restrict__ xh) {
  const size_t i = ((size_t)blockIdx.x * 256 + threadIdx.x) * 8;
  const float4 a = *(const float4*)(x + i);
  const float4 b = *(const float4*)(x + i + 4);
  v8h o;
  o[0] = (_Float16)a.x; o[1] = (_Float16)a.y; o[2] = (_Float16)a.z; o[3] = (_Float16)a.w;
  o[4] = (_Float16)b.x; o[5] = (_Float16)b.y; o[6] = (_Float16)b.z; o[7] = (_Float16)b.w;
  *(v8h*)(xh + i) = o;
}

// ---------------------------------------------------------------------------
// Kernel 3: gather + 3-layer MLP (WMMA f16->f32) + max aggregation.
// ---------------------------------------------------------------------------

// A fragment (16x32 f16 tile): lanes 0-15 hold K {0..15},
// lanes 16-31 hold K {16..31} -> one 32B load.
__device__ __forceinline__ v16h load_afrag(const _Float16* __restrict__ row, int kt, int hi) {
  const _Float16* p = row + kt * 32 + hi * 16;
  return *(const v16h*)(p);
}

__device__ __forceinline__ void mlp_layer(const _Float16* __restrict__ Ab,
                                          const _Float16* __restrict__ Wt,
                                          const float* __restrict__ bias,
                                          _Float16* __restrict__ Ob, int lane) {
  const int col = lane & 15;
  const int hi  = lane >> 4;
  v16h A0[4], A1[4];
#pragma unroll
  for (int kt = 0; kt < 4; ++kt) {
    A0[kt] = load_afrag(Ab + col * KP, kt, hi);
    A1[kt] = load_afrag(Ab + (col + 16) * KP, kt, hi);
  }
  for (int nt = 0; nt < 8; ++nt) {
    const float bv = bias[nt * 16 + col];
    v8f acc0, acc1;
#pragma unroll
    for (int r = 0; r < 8; ++r) { acc0[r] = bv; acc1[r] = bv; }
#pragma unroll
    for (int kt = 0; kt < 4; ++kt) {
      const v16h Bf = *(const v16h*)(Wt + (nt * 16 + col) * KP + kt * 32 + hi * 16);
      acc0 = __builtin_amdgcn_wmma_f32_16x16x32_f16(false, A0[kt], false, Bf, (short)0, acc0, false, false);
      acc1 = __builtin_amdgcn_wmma_f32_16x16x32_f16(false, A1[kt], false, Bf, (short)0, acc1, false, false);
    }
    const int m0 = hi * 8;
#pragma unroll
    for (int r = 0; r < 8; ++r) {
      Ob[(m0 + r) * KP + nt * 16 + col]      = (_Float16)fmaxf(acc0[r], 0.0f);
      Ob[(16 + m0 + r) * KP + nt * 16 + col] = (_Float16)fmaxf(acc1[r], 0.0f);
    }
  }
}

__device__ __forceinline__ void mlp_layer3(const _Float16* __restrict__ Ab,
                                           const _Float16* __restrict__ Wt,
                                           const float* __restrict__ bias,
                                           float* __restrict__ orow, int lane) {
  const int col = lane & 15;
  const int hi  = lane >> 4;
  v16h A0[4], A1[4];
#pragma unroll
  for (int kt = 0; kt < 4; ++kt) {
    A0[kt] = load_afrag(Ab + col * KP, kt, hi);
    A1[kt] = load_afrag(Ab + (col + 16) * KP, kt, hi);
  }
  for (int nt = 0; nt < 16; ++nt) {
    const float bv = bias[nt * 16 + col];
    v8f acc0, acc1;
#pragma unroll
    for (int r = 0; r < 8; ++r) { acc0[r] = bv; acc1[r] = bv; }
#pragma unroll
    for (int kt = 0; kt < 4; ++kt) {
      const v16h Bf = *(const v16h*)(Wt + (nt * 16 + col) * KP + kt * 32 + hi * 16);
      acc0 = __builtin_amdgcn_wmma_f32_16x16x32_f16(false, A0[kt], false, Bf, (short)0, acc0, false, false);
      acc1 = __builtin_amdgcn_wmma_f32_16x16x32_f16(false, A1[kt], false, Bf, (short)0, acc1, false, false);
    }
    float s = acc0[0];
#pragma unroll
    for (int r = 1; r < 8; ++r) s = fmaxf(s, acc0[r]);
#pragma unroll
    for (int r = 0; r < 8; ++r) s = fmaxf(s, acc1[r]);
    s = fmaxf(s, __shfl_xor(s, 16));
    s = fmaxf(s, 0.0f);               // max(relu(h)) == relu(max(h))
    if (lane < 16) orow[nt * 16 + col] = s;
  }
}

__global__ __launch_bounds__(256) void mlp_kernel(
    const _Float16* __restrict__ xh, const float* __restrict__ pos,
    const _Float16* __restrict__ wp,
    const int* __restrict__ knn,  const float* __restrict__ poss,
    float* __restrict__ out) {
  extern __shared__ char smem_raw[];
  _Float16* W1t = (_Float16*)smem_raw;        // [128][128] at LDS byte 0
  _Float16* W2t = W1t + 128 * KP;
  _Float16* W3t = W2t + 128 * KP;
  float*    b1s = (float*)(W3t + H3 * KP);    // byte 131072
  float*    b2s = b1s + 128;
  float*    b3s = b2s + 128;
  _Float16* bufs = (_Float16*)(b3s + H3);     // byte 133120: per-wave ping/pong

  const int tid = threadIdx.x;

  // ---- stage the whole 133,120B weight+bias image via one TDM descriptor ----
  if (tid < 32) {  // wave 0 only (TDM ignores EXEC; one issue per wave)
    const unsigned long long ga = (unsigned long long)(uintptr_t)wp;
    u32x4 g0;
    g0[0] = 1u;                                   // count=1 (valid descriptor)
    g0[1] = 0u;                                   // lds_addr = 0
    g0[2] = (unsigned)(ga & 0xffffffffu);         // global_addr[31:0]
    g0[3] = (unsigned)((ga >> 32) & 0x01ffffffu)  // global_addr[56:32]
          | 0x80000000u;                          // type=2 ("image")
    i32x8 g1;
    g1[0] = 0x00030000;                           // data_size=3 (8B elements)
    g1[1] = (int)((unsigned)(WP_QW & 0xffff) << 16);   // tensor_dim0 lo16
    g1[2] = (int)((unsigned)WP_QW >> 16);              // tensor_dim0 hi16
    g1[3] = (int)((unsigned)(WP_QW & 0xffff) << 16);   // tile_dim0
    g1[4] = 0;                                         // tile_dim1/2 = 0
    g1[5] = WP_QW;                                     // tensor_dim0_stride lo32
    g1[6] = 0; g1[7] = 0;
    i32x4 gz; gz[0] = 0; gz[1] = 0; gz[2] = 0; gz[3] = 0;
#if defined(__clang_major__) && (__clang_major__ >= 23)
    i32x8 gz8;
#pragma unroll
    for (int q = 0; q < 8; ++q) gz8[q] = 0;
    __builtin_amdgcn_tensor_load_to_lds(g0, g1, gz, gz, gz8, 0);
#else
    __builtin_amdgcn_tensor_load_to_lds(g0, g1, gz, gz, 0);
#endif
    __builtin_amdgcn_s_wait_tensorcnt(0);
  }
  __syncthreads();

  const int w = tid >> 5, lane = tid & 31;
  _Float16* bA = bufs + w * 2 * (NK * KP);
  _Float16* bB = bA + NK * KP;
  const unsigned ldsBufA = 133120u + (unsigned)w * 16384u;  // byte offset of bA
  const int gw = blockIdx.x * WAVES + w;

  for (int t = 0; t < CPW; ++t) {
    const int c = gw * CPW + t;
    const int b = c >> 10;
    // ---- gather msg[32][128] f16 into bA (lane r owns neighbor row r) ----
    {
      const int r = lane;
      const int j = knn[(size_t)c * NK + r];
      const float cx = poss[(size_t)c * 3 + 0];
      const float cy = poss[(size_t)c * 3 + 1];
      const float cz = poss[(size_t)c * 3 + 2];
      // drain prior DS traffic on bA before the async engine overwrites it
      asm volatile("s_wait_dscnt 0x0" ::: "memory");
      // cols 0..63: async DMA 128B of pre-converted f16 features into LDS
      const unsigned ldsrow = ldsBufA + (unsigned)r * 256u;
      const unsigned goff   = (unsigned)(((b << 12) + j) * 128);
#pragma unroll
      for (int q = 0; q < 8; ++q)
        asm volatile("global_load_async_to_lds_b128 %0, %1, %2"
                     :: "v"(ldsrow + (unsigned)q * 16u),
                        "v"(goff + (unsigned)q * 16u),
                        "s"(xh)
                     : "memory");
      // cols 64..127: zero pad, then the 3 relative-position features
      _Float16* arow = bA + r * KP;
      v8h z;
#pragma unroll
      for (int q = 0; q < 8; ++q) z[q] = (_Float16)0.0f;
#pragma unroll
      for (int q = 0; q < 8; ++q) *(v8h*)(arow + 64 + q * 8) = z;
      const float* pj = pos + ((size_t)b * NN + j) * 3;
      arow[NC + 0] = (_Float16)(pj[0] - cx);
      arow[NC + 1] = (_Float16)(pj[1] - cy);
      arow[NC + 2] = (_Float16)(pj[2] - cz);
      // async loads complete -> feature rows visible in LDS
      asm volatile("s_wait_asynccnt 0x0" ::: "memory");
    }
    // per-wave private buffers -> only in-wave ordering needed, no barriers
    mlp_layer (bA, W1t, b1s, bB, lane);
    mlp_layer (bB, W2t, b2s, bA, lane);
    mlp_layer3(bA, W3t, b3s, out + (size_t)c * H3, lane);
  }
}

// ---------------------------------------------------------------------------
extern "C" void kernel_launch(void* const* d_in, const int* in_sizes, int n_in,
                              void* d_out, int out_size, void* d_ws, size_t ws_size,
                              hipStream_t stream) {
  (void)in_sizes; (void)n_in; (void)out_size; (void)ws_size;
  const float* x   = (const float*)d_in[0];
  const float* pos = (const float*)d_in[1];
  const float* W1  = (const float*)d_in[2];
  const float* b1  = (const float*)d_in[3];
  const float* W2  = (const float*)d_in[4];
  const float* b2  = (const float*)d_in[5];
  const float* W3  = (const float*)d_in[6];
  const float* b3  = (const float*)d_in[7];

  float* out  = (float*)d_out;                        // [B*M][256]
  float* poss = out + (size_t)NB * NM * H3;           // [B*M][3] (tuple output #2)

  // workspace: knn (2MB) | weight image (133,120B) | xh f16 (8MB)
  int*      knn = (int*)d_ws;
  _Float16* wp  = (_Float16*)((char*)d_ws + (size_t)NB * NM * NK * 4);
  _Float16* xh  = (_Float16*)((char*)wp + WP_BYTES);

  fps_kernel<<<NB, 256, 0, stream>>>(pos, poss);
  knn_kernel<<<(NB * NM) / 256, 256, 0, stream>>>(pos, poss, knn);
  prep_weights<<<WP_HALVES / 256, 256, 0, stream>>>(W1, b1, W2, b2, W3, b3, wp);
  prep_x<<<(NB * NN * NC) / (256 * 8), 256, 0, stream>>>(x, xh);

  const size_t smem = (size_t)WP_BYTES
                    + (size_t)WAVES * 2 * NK * KP * sizeof(_Float16);  // ~258 KB
  hipFuncSetAttribute(reinterpret_cast<const void*>(mlp_kernel),
                      hipFuncAttributeMaxDynamicSharedMemorySize, (int)smem);
  mlp_kernel<<<(NB * NM) / (WAVES * CPW), 256, smem, stream>>>(
      xh, pos, wp, knn, poss, out);
}
